// MultiHeadAttentionSelf_81561428951219
// MI455X (gfx1250) — matrix operand order repaired and checked by reading, not compiled, not run on hardware
//
#include <hip/hip_runtime.h>

typedef _Float16 f16;
typedef __attribute__((ext_vector_type(16))) _Float16 v16h;
typedef __attribute__((ext_vector_type(8)))  _Float16 v8h;
typedef __attribute__((ext_vector_type(8)))  float    v8f;

union V16 { v16h v; v8h h[2]; };

constexpr int B_ = 8, T_ = 1024, D_ = 1280, H_ = 20, DH_ = 64;
constexpr float SCALE_ = 0.35355339059327373f;   // DH^-0.25, folded into both q and k
constexpr float NEG_INF_ = -3.0e38f;

// ---- CDNA5 async copy (global -> LDS), ASYNCcnt-tracked ----
// VDST = 32-bit LDS byte offset, VADDR = 64-bit global address, SADDR = NULL ("off").
static __device__ __forceinline__ void async_load_b128(const f16* gsrc, f16* ldst) {
  unsigned loff = (unsigned)(unsigned long long)(void*)ldst;   // low 32 bits = LDS offset
  asm volatile("global_load_async_to_lds_b128 %0, %1, off"
               :: "v"(loff), "v"(gsrc) : "memory");
}

#if __has_builtin(__builtin_amdgcn_s_wait_asynccnt)
#define WAIT_ASYNC(n) __builtin_amdgcn_s_wait_asynccnt(n)
#else
#define WAIT_ASYNC(n) asm volatile("s_wait_asynccnt %0" :: "i"(n) : "memory")
#endif

static __device__ __forceinline__ v8f wmma_f16(v16h a, v16h b, v8f c) {
  // D = A(16x32 f16) * B(32x16 f16) + C(16x16 f32)
  return __builtin_amdgcn_wmma_f32_16x16x32_f16(false, a, false, b, (short)0, c, false, false);
}

static __device__ __forceinline__ v16h load_frag_2x8(const f16* p0, const f16* p1) {
  V16 u;
  u.h[0] = *(const v8h*)p0;
  u.h[1] = *(const v8h*)p1;
  return u.v;
}

// ---------------- prep kernels ----------------

__global__ void cvt_f32_to_f16(const float* __restrict__ src, f16* __restrict__ dst, int n) {
  int i = blockIdx.x * blockDim.x + threadIdx.x;
  if (i < n) dst[i] = (f16)src[i];
}

// W [k][n] f32 (row-major) -> WT [n][k] f16
__global__ void transpose_to_f16(const float* __restrict__ W, f16* __restrict__ WT, int dim) {
  int i = blockIdx.x * blockDim.x + threadIdx.x;
  if (i < dim * dim) {
    int n = i / dim, k = i - n * dim;
    WT[i] = (f16)W[(size_t)k * dim + n];
  }
}

// ---------------- fused QKV projection ----------------
// grid (B*T/64, D/64, 3); block 128 (4 waves).
// WG tile 64x64: wave w owns rows m0+16w..+15; B tile (64 cols x 32 K) is
// async-copied to LDS once per WG and double-buffered across a fully uniform
// K loop (wrap-around prefetch keeps the latch single-path for phi coalescing).
__global__ __launch_bounds__(128) void gemm_qkv_kernel(
    const f16* __restrict__ xh,
    const f16* __restrict__ WqT, const f16* __restrict__ WkT, const f16* __restrict__ WvT,
    const float* __restrict__ bq, const float* __restrict__ bv,
    float* __restrict__ kout, float* __restrict__ vout,
    f16* __restrict__ qh, f16* __restrict__ kh, f16* __restrict__ vt)
{
  __shared__ f16 ldsB[2][64 * 32];          // [buf][n_local*32 + k_local]
  const int tid  = threadIdx.x;
  const int lane = tid & 31;
  const int wid  = tid >> 5;
  const int l    = lane & 15;
  const bool hi  = lane >= 16;
  const int z    = blockIdx.z;              // 0=Q, 1=K, 2=V
  const int m0   = blockIdx.x * 64 + wid * 16;
  const int n0   = blockIdx.y * 64;

  const f16* WT = (z == 0) ? WqT : (z == 1) ? WkT : WvT;

  // each thread async-copies two 16B chunks per buffer (256 chunks = 4KB tile)
  auto prefetch = [&](int kb, int buf) {
#pragma unroll
    for (int c = 0; c < 2; ++c) {
      const int ch  = tid + 128 * c;
      const int row = ch >> 2;              // 0..63  (n_local)
      const int q   = ch & 3;               // 16B chunk within the 64B row
      async_load_b128(WT + (size_t)(n0 + row) * D_ + kb + q * 8,
                      &ldsB[buf][row * 32 + q * 8]);
    }
  };

  v8f acc[4];
#pragma unroll
  for (int n = 0; n < 4; ++n)
#pragma unroll
    for (int j = 0; j < 8; ++j) acc[n][j] = 0.0f;

  const f16* arow  = xh + (size_t)(m0 + l) * D_;
  const int ka_off = hi ? 8 : 0;

  prefetch(0, 0);
  v16h a_cur = load_frag_2x8(arow + ka_off, arow + ka_off + 16);

  constexpr int NIT = D_ / 32;              // 40
#pragma unroll 2
  for (int it = 0; it < NIT; ++it) {
    const int buf     = it & 1;
    const int kb      = it * 32;
    const int kb_next = (kb + 32 < D_) ? kb + 32 : 0;   // wrap: uniform latch

    prefetch(kb_next, buf ^ 1);             // next tile in flight behind WMMA
    WAIT_ASYNC(2);                          // only the current buffer must land
    __syncthreads();

    const f16* an = arow + kb_next + ka_off;
    v16h a_next = load_frag_2x8(an, an + 16);   // register-pipelined A

    // preload all 4 B fragments, then issue 4 back-to-back WMMAs
    const f16* bbase = &ldsB[buf][hi ? 8 : 0];
    v16h bf[4];
#pragma unroll
    for (int nt = 0; nt < 4; ++nt) {
      const f16* br = bbase + (16 * nt + l) * 32;
      bf[nt] = load_frag_2x8(br, br + 16);
    }
#pragma unroll
    for (int nt = 0; nt < 4; ++nt)
      acc[nt] = wmma_f16(a_cur, bf[nt], acc[nt]);

    a_cur = a_next;
    __syncthreads();                        // buffer reuse fence
  }
  // trailing wrap-around prefetch drains via S_ENDPGM's implicit wait-idle

#pragma unroll
  for (int nt = 0; nt < 4; ++nt) {
    const int col = n0 + 16 * nt + l;
    const int h   = col >> 6;               // / DH_
    const int dh  = col & 63;
#pragma unroll
    for (int j = 0; j < 8; ++j) {
      const int r = m0 + j + (hi ? 8 : 0);
      const int b = r >> 10;                // / T_
      const int t = r & 1023;
      const float v = acc[nt][j];
      if (z == 0) {
        const float q = v + bq[col];
        qh[(((size_t)b * H_ + h) * T_ + t) * DH_ + dh] = (f16)(q * SCALE_);
      } else if (z == 1) {
        kout[(size_t)r * D_ + col] = v;     // key proj: no bias
        kh[(((size_t)b * H_ + h) * T_ + t) * DH_ + dh] = (f16)(v * SCALE_);
      } else {
        const float vv = v + bv[col];
        vout[(size_t)r * D_ + col] = vv;
        vt[(((size_t)b * H_ + h) * DH_ + dh) * T_ + t] = (f16)vv;  // transposed for PV
      }
    }
  }
}

// ---------------- flash attention ----------------
// grid (B*H, T/64); block 128 (4 waves); wave handles 16 queries, full DH=64.
__global__ __launch_bounds__(128) void attn_kernel(
    const f16* __restrict__ qh, const f16* __restrict__ kh,
    const f16* __restrict__ vt, f16* __restrict__ wvh)
{
  __shared__ f16 lds_p[4][16][32];
  const int lane = threadIdx.x & 31;
  const int wid  = threadIdx.x >> 5;
  const int l    = lane & 15;
  const bool hi  = lane >= 16;
  const int bh   = blockIdx.x;
  const int b    = bh / H_;
  const int h    = bh - b * H_;
  const int q0   = (blockIdx.y * 4 + wid) * 16;

  const f16* Q = qh + (size_t)bh * T_ * DH_;
  const f16* K = kh + (size_t)bh * T_ * DH_;
  const f16* V = vt + (size_t)bh * DH_ * T_;

  // Q A-fragments for k-dim 0..31 and 32..63 (resident for the whole loop)
  v16h qa[2];
  {
    const f16* qr = Q + (size_t)(q0 + l) * DH_;
#pragma unroll
    for (int p = 0; p < 2; ++p) {
      const int o = 32 * p + (hi ? 8 : 0);
      qa[p] = load_frag_2x8(qr + o, qr + o + 16);
    }
  }

  v8f acc[4];
  float mrow[8], lrow[8];
#pragma unroll
  for (int j = 0; j < 8; ++j) { mrow[j] = NEG_INF_; lrow[j] = 0.0f; }
#pragma unroll
  for (int n = 0; n < 4; ++n)
#pragma unroll
    for (int j = 0; j < 8; ++j) acc[n][j] = 0.0f;

  const int kend = q0 + 16;                 // causal: keys < q0+16
  for (int kb = 0; kb < kend; kb += 32) {
    // issue ALL loads for this key tile up front: 4 K-fragments + 4 V-fragments.
    // V loads overlap the S WMMAs and the softmax VALU work below.
    v16h kf[2][2];
#pragma unroll
    for (int g = 0; g < 2; ++g) {
      const f16* kr = K + (size_t)(kb + 16 * g + l) * DH_ + (hi ? 8 : 0);
#pragma unroll
      for (int p = 0; p < 2; ++p)
        kf[g][p] = load_frag_2x8(kr + 32 * p, kr + 32 * p + 16);
    }
    v16h vf[4];
#pragma unroll
    for (int n = 0; n < 4; ++n) {
      const f16* vr = V + (size_t)(16 * n + l) * T_ + kb + (hi ? 8 : 0);
      vf[n] = load_frag_2x8(vr, vr + 16);
    }

    // S = Q * K^T : two 16-key tiles, each 2 WMMA over DH=64
    v8f s[2];
#pragma unroll
    for (int g = 0; g < 2; ++g) {
#pragma unroll
      for (int j = 0; j < 8; ++j) s[g][j] = 0.0f;
#pragma unroll
      for (int p = 0; p < 2; ++p)
        s[g] = wmma_f16(qa[p], kf[g][p], s[g]);
    }

    // causal mask (C-layout: lane's column is kb[+16]+l, rows j/(j+8))
#pragma unroll
    for (int j = 0; j < 8; ++j) {
      const int row = q0 + j + (hi ? 8 : 0);
      if (kb + l      > row) s[0][j] = NEG_INF_;
      if (kb + 16 + l > row) s[1][j] = NEG_INF_;
    }

    // online softmax; row stats reduced across the 16-lane half sharing a row group
#pragma unroll
    for (int j = 0; j < 8; ++j) {
      float t = fmaxf(s[0][j], s[1][j]);
      t = fmaxf(t, __shfl_xor(t, 1));
      t = fmaxf(t, __shfl_xor(t, 2));
      t = fmaxf(t, __shfl_xor(t, 4));
      t = fmaxf(t, __shfl_xor(t, 8));
      const float mnew  = fmaxf(mrow[j], t);
      const float alpha = __expf(mrow[j] - mnew);
      const float p0 = __expf(s[0][j] - mnew);
      const float p1 = __expf(s[1][j] - mnew);
      float ps = p0 + p1;
      ps += __shfl_xor(ps, 1);
      ps += __shfl_xor(ps, 2);
      ps += __shfl_xor(ps, 4);
      ps += __shfl_xor(ps, 8);
      lrow[j] = lrow[j] * alpha + ps;
      mrow[j] = mnew;
#pragma unroll
      for (int n = 0; n < 4; ++n) acc[n][j] *= alpha;
      const int prow = j + (hi ? 8 : 0);
      lds_p[wid][prow][l]      = (f16)p0;   // C-layout -> LDS tile
      lds_p[wid][prow][16 + l] = (f16)p1;
    }

    __builtin_amdgcn_wave_barrier();        // same-wave DS ops are in-order

    // re-read P in A-matrix 16x32 layout
    const f16* pr = &lds_p[wid][l][0] + (hi ? 8 : 0);
    v16h pA = load_frag_2x8(pr, pr + 16);

    __builtin_amdgcn_wave_barrier();

    // acc += P(16x32) * V(32x64): 4 back-to-back WMMAs with preloaded V
#pragma unroll
    for (int n = 0; n < 4; ++n)
      acc[n] = wmma_f16(pA, vf[n], acc[n]);
  }

  // normalize and emit wv in [b][t][d] f16 for the output projection
  float rcp[8];
#pragma unroll
  for (int j = 0; j < 8; ++j) rcp[j] = 1.0f / lrow[j];
#pragma unroll
  for (int n = 0; n < 4; ++n) {
    const int d = h * DH_ + 16 * n + l;
#pragma unroll
    for (int j = 0; j < 8; ++j) {
      const int row = q0 + j + (hi ? 8 : 0);
      wvh[((size_t)b * T_ + row) * D_ + d] = (f16)(acc[n][j] * rcp[j]);
    }
  }
}

// ---------------- output projection ----------------
// Same async-LDS double-buffered uniform-latch tiling as the QKV GEMM.
__global__ __launch_bounds__(128) void gemm_out_kernel(
    const f16* __restrict__ wvh, const f16* __restrict__ WoT,
    const float* __restrict__ bo, float* __restrict__ out)
{
  __shared__ f16 ldsB[2][64 * 32];
  const int tid  = threadIdx.x;
  const int lane = tid & 31;
  const int wid  = tid >> 5;
  const int l    = lane & 15;
  const bool hi  = lane >= 16;
  const int m0   = blockIdx.x * 64 + wid * 16;
  const int n0   = blockIdx.y * 64;

  auto prefetch = [&](int kb, int buf) {
#pragma unroll
    for (int c = 0; c < 2; ++c) {
      const int ch  = tid + 128 * c;
      const int row = ch >> 2;
      const int q   = ch & 3;
      async_load_b128(WoT + (size_t)(n0 + row) * D_ + kb + q * 8,
                      &ldsB[buf][row * 32 + q * 8]);
    }
  };

  v8f acc[4];
#pragma unroll
  for (int n = 0; n < 4; ++n)
#pragma unroll
    for (int j = 0; j < 8; ++j) acc[n][j] = 0.0f;

  const f16* arow  = wvh + (size_t)(m0 + l) * D_;
  const int ka_off = hi ? 8 : 0;

  prefetch(0, 0);
  v16h a_cur = load_frag_2x8(arow + ka_off, arow + ka_off + 16);

  constexpr int NIT = D_ / 32;
#pragma unroll 2
  for (int it = 0; it < NIT; ++it) {
    const int buf     = it & 1;
    const int kb      = it * 32;
    const int kb_next = (kb + 32 < D_) ? kb + 32 : 0;

    prefetch(kb_next, buf ^ 1);
    WAIT_ASYNC(2);
    __syncthreads();

    const f16* an = arow + kb_next + ka_off;
    v16h a_next = load_frag_2x8(an, an + 16);

    const f16* bbase = &ldsB[buf][hi ? 8 : 0];
    v16h bf[4];
#pragma unroll
    for (int nt = 0; nt < 4; ++nt) {
      const f16* br = bbase + (16 * nt + l) * 32;
      bf[nt] = load_frag_2x8(br, br + 16);
    }
#pragma unroll
    for (int nt = 0; nt < 4; ++nt)
      acc[nt] = wmma_f16(a_cur, bf[nt], acc[nt]);

    a_cur = a_next;
    __syncthreads();
  }

#pragma unroll
  for (int nt = 0; nt < 4; ++nt) {
    const int col = n0 + 16 * nt + l;
#pragma unroll
    for (int j = 0; j < 8; ++j) {
      const int r = m0 + j + (hi ? 8 : 0);
      out[(size_t)r * D_ + col] = acc[nt][j] + bo[col];
    }
  }
}

// ---------------- launch ----------------
extern "C" void kernel_launch(void* const* d_in, const int* in_sizes, int n_in,
                              void* d_out, int out_size, void* d_ws, size_t ws_size,
                              hipStream_t stream) {
  (void)in_sizes; (void)n_in; (void)out_size; (void)ws_size;
  const float* x  = (const float*)d_in[0];
  // d_in[1] k_cache, d_in[2] v_cache (fully overwritten), d_in[3] mask (rebuilt): unused
  const float* Wq = (const float*)d_in[4];
  const float* bq = (const float*)d_in[5];
  const float* Wk = (const float*)d_in[6];
  const float* Wv = (const float*)d_in[7];
  const float* bv = (const float*)d_in[8];
  const float* Wo = (const float*)d_in[9];
  const float* bo = (const float*)d_in[10];

  float* out  = (float*)d_out;
  float* kout = out  + (size_t)B_ * T_ * D_;
  float* vout = kout + (size_t)B_ * T_ * D_;

  char* ws = (char*)d_ws;
  const size_t nx = (size_t)B_ * T_ * D_;
  const size_t nw = (size_t)D_ * D_;
  f16* xh  = (f16*)ws; ws += nx * sizeof(f16);
  f16* WqT = (f16*)ws; ws += nw * sizeof(f16);
  f16* WkT = (f16*)ws; ws += nw * sizeof(f16);
  f16* WvT = (f16*)ws; ws += nw * sizeof(f16);
  f16* WoT = (f16*)ws; ws += nw * sizeof(f16);
  f16* qh  = (f16*)ws; ws += nx * sizeof(f16);
  f16* kh  = (f16*)ws; ws += nx * sizeof(f16);
  f16* vt  = (f16*)ws; ws += nx * sizeof(f16);
  f16* wvh = (f16*)ws; ws += nx * sizeof(f16);

  cvt_f32_to_f16<<<dim3((unsigned)((nx + 255) / 256)), 256, 0, stream>>>(x, xh, (int)nx);
  transpose_to_f16<<<dim3((unsigned)((nw + 255) / 256)), 256, 0, stream>>>(Wq, WqT, D_);
  transpose_to_f16<<<dim3((unsigned)((nw + 255) / 256)), 256, 0, stream>>>(Wk, WkT, D_);
  transpose_to_f16<<<dim3((unsigned)((nw + 255) / 256)), 256, 0, stream>>>(Wv, WvT, D_);
  transpose_to_f16<<<dim3((unsigned)((nw + 255) / 256)), 256, 0, stream>>>(Wo, WoT, D_);

  gemm_qkv_kernel<<<dim3(B_ * T_ / 64, D_ / 64, 3), 128, 0, stream>>>(
      xh, WqT, WkT, WvT, bq, bv, kout, vout, qh, kh, vt);

  attn_kernel<<<dim3(B_ * H_, T_ / 64), 128, 0, stream>>>(qh, kh, vt, wvh);

  gemm_out_kernel<<<dim3(B_ * T_ / 64, D_ / 64), 128, 0, stream>>>(wvh, WoT, bo, out);
}
